// GraphNetworkSimulator_24558622998798
// MI455X (gfx1250) — matrix operand order repaired and checked by reading, not compiled, not run on hardware
//
#include <hip/hip_runtime.h>

typedef _Float16 h16;
typedef __attribute__((ext_vector_type(8)))  _Float16 v8h;
typedef __attribute__((ext_vector_type(16))) _Float16 v16h;
typedef __attribute__((ext_vector_type(8)))  float    v8f;

typedef __attribute__((ext_vector_type(4))) unsigned int u32x4;
typedef __attribute__((ext_vector_type(8))) int          i32x8;
typedef __attribute__((ext_vector_type(4))) int          i32x4;

#define LATENT 192
#define NTILES 12   /* 192 / 16 */
#define NHALF  6    /* n-tiles per column-half */

// ---------------------------------------------------------------------------
// Pack W (f32, K x 192 row-major) into f16 B-fragment order:
//   Wp[((kc*12 + nt)*32 + lane)*16 + j] = W[kc*32 + (lane>=16?16:0) + j][nt*16 + (lane&15)]
// Contiguous k order (lane half h holds k 16h..16h+15), as the A loads use: each lane's
// v16h operand is one contiguous 32-byte read.
// ---------------------------------------------------------------------------
__global__ void pack_w_kernel(const float* __restrict__ W, h16* __restrict__ Wp,
                              int Kreal, int total) {
  int idx = blockIdx.x * blockDim.x + threadIdx.x;
  if (idx >= total) return;
  int j  = idx & 15;
  int l  = (idx >> 4) & 31;
  int nt = (idx >> 9) % NTILES;
  int kc = idx / (NTILES * 512);
  int k  = kc * 32 + ((l >> 4) << 4) + j;
  int n  = nt * 16 + (l & 15);
  float v = (k < Kreal) ? W[(size_t)k * LATENT + n] : 0.0f;
  Wp[idx] = (h16)v;
}

__global__ void zero_kernel(float* __restrict__ p, int n) {
  int i = blockIdx.x * blockDim.x + threadIdx.x;
  if (i < n) p[i] = 0.0f;
}

// Pad a M x D f32 matrix into M x 32 f16 (zero-padded K for the K=32 GEMM).
__global__ void pad_cast_kernel(const float* __restrict__ src, h16* __restrict__ dst,
                                int M, int D) {
  int idx = blockIdx.x * blockDim.x + threadIdx.x;
  if (idx >= M * 32) return;
  int r = idx >> 5, c = idx & 31;
  dst[idx] = (h16)((c < D) ? src[(size_t)r * D + c] : 0.0f);
}

// e_in = [h_e | h_n[senders] | h_n[receivers]]  (f16, E x 576)
__global__ void build_ein_kernel(const h16* __restrict__ he, const h16* __restrict__ hn,
                                 const int* __restrict__ snd, const int* __restrict__ rcv,
                                 h16* __restrict__ out) {
  int e = blockIdx.x, c = threadIdx.x;
  h16 v;
  if (c < LATENT)            v = he[(size_t)e * LATENT + c];
  else if (c < 2 * LATENT)   v = hn[(size_t)snd[e] * LATENT + (c - LATENT)];
  else                       v = hn[(size_t)rcv[e] * LATENT + (c - 2 * LATENT)];
  out[(size_t)e * (3 * LATENT) + c] = v;
}

// n_in = [h_n | sent | recv]  (f16, N x 576)
__global__ void build_nin_kernel(const h16* __restrict__ hn, const float* __restrict__ sent,
                                 const float* __restrict__ recv, h16* __restrict__ out) {
  int n = blockIdx.x, c = threadIdx.x;
  h16 v;
  if (c < LATENT)            v = hn[(size_t)n * LATENT + c];
  else if (c < 2 * LATENT)   v = (h16)sent[(size_t)n * LATENT + (c - LATENT)];
  else                       v = (h16)recv[(size_t)n * LATENT + (c - 2 * LATENT)];
  out[(size_t)n * (3 * LATENT) + c] = v;
}

// ---------------------------------------------------------------------------
// TDM: DMA one 6144-byte weight half-slab (32 K-rows x 96 cols, f16, already
// fragment-packed contiguously) from global into LDS. 1-row tile of 768
// 8-byte elements. D# per CDNA5 ISA ch.8; tracked by TENSORcnt.
// (6-arg builtin form: g0 u32x4, g1 i32x8, g2 i32x4, g3 i32x4, i32x8, cpol)
// ---------------------------------------------------------------------------
__device__ __forceinline__ void tdm_load_slab(const h16* gsrc, h16* ldsDst) {
  uint64_t ga = (uint64_t)(uintptr_t)gsrc;
  uint32_t la = (uint32_t)(uintptr_t)ldsDst;  // low 32 bits of generic = LDS offset
  u32x4 g0 = { 1u,                                  // count=1 (valid user D#)
               la,                                  // lds_addr      [63:32]
               (uint32_t)ga,                        // global_addr   [95:64]
               (uint32_t)(ga >> 32) | (2u << 30) }; // ga[56:32] + type=2 [127:126]
  i32x8 g1 = { (int)(3u << 16),    // workgroup_mask=0, data_size=3 (8B)
               (int)(768u << 16),  // tensor_dim0 lo16 in [63:48]
               (int)(1u << 16),    // tensor_dim0 hi16=0, tensor_dim1 lo16=1
               (int)(768u << 16),  // tensor_dim1 hi16=0, tile_dim0=768
               1,                  // tile_dim1=1, tile_dim2=0
               768,                // tensor_dim0_stride lo32
               0, 0 };
  i32x4 gz4 = { 0, 0, 0, 0 };
  i32x8 gz8 = { 0, 0, 0, 0, 0, 0, 0, 0 };
  __builtin_amdgcn_tensor_load_to_lds(g0, g1, gz4, gz4, gz8, 0);
}

// ---------------------------------------------------------------------------
// WMMA GEMM: Y(M x 192) = X(M x 32*KC f16 row-major) * Wpacked + bias.
// Grid: (ceil(M/256), 2). Block = 256 threads = 8 waves; each wave computes a
// 32-row x 96-col strip (two A fragments reuse each B fragment -> half the
// LDS traffic per FLOP); blockIdx.y picks the 96-column half.
// Weight half-slabs (6KB) are TDM-DMA'd into double-buffered LDS: thread 0
// issues tensor_load_to_lds for chunk kc+1 while all waves run 12 WMMAs on
// chunk kc; one s_wait_tensorcnt + barrier per chunk.
// MODE 0: ReLU -> f16 out (hidden layers)
// MODE 1: linear -> f32 + f16 out (encoder final layer)
// MODE 2: linear, residual out32 += v (f32+f16), scatter-add v to sent/recv
// MODE 3: linear, residual out32 += v (f32+f16)           (node final layer)
// ---------------------------------------------------------------------------
template<int KC, int MODE>
__global__ __launch_bounds__(256)
void gemm_wmma_kernel(const h16* __restrict__ X, const h16* __restrict__ Wp,
                      const float* __restrict__ bias, int M,
                      h16* __restrict__ out16, float* __restrict__ out32,
                      float* __restrict__ sent, float* __restrict__ recv,
                      const int* __restrict__ snd, const int* __restrict__ rcv) {
  __shared__ __attribute__((aligned(32))) h16 ldsW[2][NHALF * 512];  // 2 x 6KB
  const int lane    = threadIdx.x & 31;
  const int wave    = threadIdx.x >> 5;
  const int nh      = blockIdx.y;                 // column half (0 or 1)
  const int rowBase = (blockIdx.x * 8 + wave) * 32;
  const h16* wsrc   = Wp + (size_t)nh * (NHALF * 512);  // half-slab within chunk

  v8f acc0[NHALF] = {};
  v8f acc1[NHALF] = {};

  int ar0 = rowBase + (lane & 15);       if (ar0 >= M) ar0 = M - 1;
  int ar1 = rowBase + 16 + (lane & 15);  if (ar1 >= M) ar1 = M - 1;
  const int K = KC * 32;
  const h16* aB0 = X + (size_t)ar0 * K + ((lane >> 4) << 4);
  const h16* aB1 = X + (size_t)ar1 * K + ((lane >> 4) << 4);

  if (threadIdx.x == 0) tdm_load_slab(wsrc, &ldsW[0][0]);
  __builtin_amdgcn_s_wait_tensorcnt(0);
  __syncthreads();

  for (int kc = 0; kc < KC; ++kc) {
    if (kc + 1 < KC && threadIdx.x == 0)
      tdm_load_slab(wsrc + (size_t)(kc + 1) * (NTILES * 512), &ldsW[(kc + 1) & 1][0]);

    // A fragments: lanes 0-15 hold K 0-15, lanes 16-31 hold K 16-31
    v8h p0 = *reinterpret_cast<const v8h*>(aB0 + kc * 32);
    v8h p1 = *reinterpret_cast<const v8h*>(aB0 + kc * 32 + 8);
    v16h a0 = __builtin_shufflevector(p0, p1, 0,1,2,3,4,5,6,7,8,9,10,11,12,13,14,15);
    v8h q0 = *reinterpret_cast<const v8h*>(aB1 + kc * 32);
    v8h q1 = *reinterpret_cast<const v8h*>(aB1 + kc * 32 + 8);
    v16h a1 = __builtin_shufflevector(q0, q1, 0,1,2,3,4,5,6,7,8,9,10,11,12,13,14,15);

    const v16h* bp = reinterpret_cast<const v16h*>(&ldsW[kc & 1][0]);
#pragma unroll
    for (int nt = 0; nt < NHALF; ++nt) {
      v16h b = bp[nt * 32 + lane];
      acc0[nt] = __builtin_amdgcn_wmma_f32_16x16x32_f16(
          false, a0, false, b, (short)0, acc0[nt], false, false);
      acc1[nt] = __builtin_amdgcn_wmma_f32_16x16x32_f16(
          false, a1, false, b, (short)0, acc1[nt], false, false);
    }

    if (kc + 1 < KC) {
      __builtin_amdgcn_s_wait_tensorcnt(0);
      __syncthreads();
    }
  }

  // Epilogue. C layout: VGPR r -> row r (+8 for lanes 16-31), col = lane&15.
  const int col0   = lane & 15;
  const int rowOff = (lane >> 4) << 3;
  float bcol[NHALF];
#pragma unroll
  for (int nt = 0; nt < NHALF; ++nt) bcol[nt] = bias[(nh * NHALF + nt) * 16 + col0];

#pragma unroll
  for (int g = 0; g < 2; ++g) {
    const v8f* accg = g ? acc1 : acc0;
#pragma unroll
    for (int r = 0; r < 8; ++r) {
      int R = rowBase + g * 16 + rowOff + r;
      if (R >= M) continue;
      size_t rowIdx = (size_t)R * LATENT;
      int se = 0, re = 0;
      if constexpr (MODE == 2) { se = snd[R]; re = rcv[R]; }
#pragma unroll
      for (int nt = 0; nt < NHALF; ++nt) {
        int c = (nh * NHALF + nt) * 16 + col0;
        float v = accg[nt][r] + bcol[nt];
        if constexpr (MODE == 0) {
          v = v > 0.0f ? v : 0.0f;
          out16[rowIdx + c] = (h16)v;
        } else if constexpr (MODE == 1) {
          out32[rowIdx + c] = v;
          out16[rowIdx + c] = (h16)v;
        } else {
          float hv = out32[rowIdx + c] + v;    // residual in fp32
          out32[rowIdx + c] = hv;
          out16[rowIdx + c] = (h16)hv;
          if constexpr (MODE == 2) {
            atomicAdd(&sent[(size_t)se * LATENT + c], v);
            atomicAdd(&recv[(size_t)re * LATENT + c], v);
          }
        }
      }
    }
  }
}

// Final decoder layer: 192 -> 1. One wave per row, 6 halves per lane,
// wave32 xor-shuffle reduction.
__global__ void dec_final_kernel(const h16* __restrict__ X, const float* __restrict__ W2,
                                 const float* __restrict__ b2, float* __restrict__ accOut,
                                 int M) {
  int row  = blockIdx.x * 8 + (threadIdx.x >> 5);
  int lane = threadIdx.x & 31;
  if (row >= M) return;
  const h16* x = X + (size_t)row * LATENT + lane * 6;
  const float* w = W2 + lane * 6;
  float s = 0.0f;
#pragma unroll
  for (int k = 0; k < 6; ++k) s += (float)x[k] * w[k];
#pragma unroll
  for (int off = 16; off > 0; off >>= 1) s += __shfl_xor(s, off, 32);
  if (lane == 0) accOut[row] = s + b2[0];   // ACC_STD=1, ACC_MEAN=0
}

// Integrator + output assembly: next_nodes (N x 8) then next_edges (N-1).
__global__ void finalize_kernel(const float* __restrict__ nodes, const float* __restrict__ next_u,
                                const float* __restrict__ acc, float* __restrict__ out, int N) {
  int r = blockIdx.x * blockDim.x + threadIdx.x;
  if (r >= N) return;
  const float h = 0.06f;  // DT * NUM_MP_STEPS
  float pa = acc[r];
  float nv = nodes[(size_t)r * 7 + 5] + pa * h;
  float np = nodes[(size_t)r * 7 + 0] + nv * h;
  float* on = out + (size_t)r * 8;
  on[0] = np;
  on[1] = nodes[(size_t)r * 7 + 2];
  on[2] = nodes[(size_t)r * 7 + 3];
  on[3] = nodes[(size_t)r * 7 + 4];
  on[4] = nodes[(size_t)r * 7 + 5];
  on[5] = nv;
  on[6] = next_u[2 * r + 1];
  on[7] = pa;
  if (r + 1 < N) {
    float pa2 = acc[r + 1];
    float nv2 = nodes[(size_t)(r + 1) * 7 + 5] + pa2 * h;
    float np2 = nodes[(size_t)(r + 1) * 7 + 0] + nv2 * h;
    out[(size_t)N * 8 + r] = np2 - np;
  }
}

// ---------------------------------------------------------------------------
extern "C" void kernel_launch(void* const* d_in, const int* in_sizes, int n_in,
                              void* d_out, int out_size, void* d_ws, size_t ws_size,
                              hipStream_t stream) {
  (void)out_size; (void)ws_size;
  if (n_in < 6) return;

  const float* nodes     = (const float*)d_in[0];
  const float* edges     = (const float*)d_in[1];
  const int*   senders   = (const int*)d_in[2];
  const int*   receivers = (const int*)d_in[3];
  const float* next_u    = (const float*)d_in[4];
  const int N = in_sizes[0] / 7;
  const int E = in_sizes[1];

  // ---- parse param leaves by size signature (order-scheme agnostic) ----
  struct Lyr { const float* W; const float* b; };
  Lyr en[2] = {}, ee[2] = {}, dc[2][3] = {}, pE[6][2] = {}, pN[6][2] = {};
  int nproc = 0, ndec = 0;
  for (int i = 5; i < n_in;) {
    int s = in_sizes[i];
    if (s == 1344) {                                    // enc_node: 7x192 MLP
      en[0] = {(const float*)d_in[i],   (const float*)d_in[i+1]};
      en[1] = {(const float*)d_in[i+2], (const float*)d_in[i+3]};
      i += 4;
    } else if (s == 110592) {                           // proc MLP (576x192,...)
      Lyr* dst2 = (nproc & 1) ? pN[nproc >> 1] : pE[nproc >> 1];
      dst2[0] = {(const float*)d_in[i],   (const float*)d_in[i+1]};
      dst2[1] = {(const float*)d_in[i+2], (const float*)d_in[i+3]};
      if (nproc < 12) nproc++;
      i += 4;
    } else if (s == 36864) {                            // decoder 3-layer MLP
      if (ndec < 2) {
        dc[ndec][0] = {(const float*)d_in[i],   (const float*)d_in[i+1]};
        dc[ndec][1] = {(const float*)d_in[i+2], (const float*)d_in[i+3]};
        dc[ndec][2] = {(const float*)d_in[i+4], (const float*)d_in[i+5]};
      }
      ndec++;
      i += 6;
    } else if (s == 192) {                              // enc_edge: 1x192 MLP
      ee[0] = {(const float*)d_in[i],   (const float*)d_in[i+1]};
      ee[1] = {(const float*)d_in[i+2], (const float*)d_in[i+3]};
      i += 4;
    } else {
      i++;
    }
  }
  // sorted pytree flatten puts dec_edge first (leaf 5 == 36864); insertion
  // order puts enc_node first (leaf 5 == 1344) with dec_node before dec_edge.
  Lyr* dn = (in_sizes[5] == 36864) ? dc[1] : dc[0];

  // ---- workspace carve ----
  size_t off = 0;
  char* base = (char*)d_ws;
  auto alloc = [&](size_t bytes) -> char* {
    char* p = base + off;
    off += (bytes + 255) & ~(size_t)255;
    return p;
  };
  float* hn32 = (float*)alloc((size_t)N * LATENT * 4);
  float* he32 = (float*)alloc((size_t)E * LATENT * 4);
  h16*   hn16 = (h16*)  alloc((size_t)N * LATENT * 2);
  h16*   he16 = (h16*)  alloc((size_t)E * LATENT * 2);
  float* sent = (float*)alloc((size_t)N * LATENT * 4);
  float* recv = (float*)alloc((size_t)N * LATENT * 4);
  h16*   in576= (h16*)  alloc((size_t)N * 3 * LATENT * 2);
  h16*   tmp16= (h16*)  alloc((size_t)N * LATENT * 2);
  h16*   xn16 = (h16*)  alloc((size_t)N * 32 * 2);
  h16*   xe16 = (h16*)  alloc((size_t)E * 32 * 2);
  float* pred = (float*)alloc((size_t)N * 4);
  h16*   tmp2 = in576;  // reuse for decoder's 2nd hidden activation

  auto allocW = [&](int KC) -> h16* { return (h16*)alloc((size_t)KC * 6144 * 2); };
  h16* pwEN0 = allocW(1);  h16* pwEN1 = allocW(6);
  h16* pwEE0 = allocW(1);  h16* pwEE1 = allocW(6);
  h16* pwD0  = allocW(6);  h16* pwD1  = allocW(6);
  h16 *pwPE0[6], *pwPE1[6], *pwPN0[6], *pwPN1[6];
  for (int s = 0; s < 6; ++s) {
    pwPE0[s] = allocW(18); pwPE1[s] = allocW(6);
    pwPN0[s] = allocW(18); pwPN1[s] = allocW(6);
  }

  auto pack = [&](const float* W, h16* Wp, int Kreal, int KC) {
    int total = KC * 6144;
    pack_w_kernel<<<(total + 255) / 256, 256, 0, stream>>>(W, Wp, Kreal, total);
  };
  auto grid = [](int M) { return dim3((unsigned)((M + 255) / 256), 2, 1); };

  // ---- pack all weights (tiny; once per launch) ----
  pack(en[0].W, pwEN0, 7, 1);    pack(en[1].W, pwEN1, 192, 6);
  pack(ee[0].W, pwEE0, 1, 1);    pack(ee[1].W, pwEE1, 192, 6);
  pack(dn[0].W, pwD0, 192, 6);   pack(dn[1].W, pwD1, 192, 6);
  for (int s = 0; s < 6; ++s) {
    pack(pE[s][0].W, pwPE0[s], 576, 18);  pack(pE[s][1].W, pwPE1[s], 192, 6);
    pack(pN[s][0].W, pwPN0[s], 576, 18);  pack(pN[s][1].W, pwPN1[s], 192, 6);
  }

  // ---- encoders ----
  pad_cast_kernel<<<(N * 32 + 255) / 256, 256, 0, stream>>>(nodes, xn16, N, 7);
  pad_cast_kernel<<<(E * 32 + 255) / 256, 256, 0, stream>>>(edges, xe16, E, 1);
  gemm_wmma_kernel<1, 0><<<grid(N), 256, 0, stream>>>(xn16, pwEN0, en[0].b, N,
      tmp16, nullptr, nullptr, nullptr, nullptr, nullptr);
  gemm_wmma_kernel<6, 1><<<grid(N), 256, 0, stream>>>(tmp16, pwEN1, en[1].b, N,
      hn16, hn32, nullptr, nullptr, nullptr, nullptr);
  gemm_wmma_kernel<1, 0><<<grid(E), 256, 0, stream>>>(xe16, pwEE0, ee[0].b, E,
      tmp16, nullptr, nullptr, nullptr, nullptr, nullptr);
  gemm_wmma_kernel<6, 1><<<grid(E), 256, 0, stream>>>(tmp16, pwEE1, ee[1].b, E,
      he16, he32, nullptr, nullptr, nullptr, nullptr);

  // ---- 6 message-passing steps ----
  const int NL = N * LATENT;
  for (int s = 0; s < 6; ++s) {
    zero_kernel<<<(NL + 255) / 256, 256, 0, stream>>>(sent, NL);
    zero_kernel<<<(NL + 255) / 256, 256, 0, stream>>>(recv, NL);

    build_ein_kernel<<<E, 3 * LATENT, 0, stream>>>(he16, hn16, senders, receivers, in576);
    gemm_wmma_kernel<18, 0><<<grid(E), 256, 0, stream>>>(in576, pwPE0[s], pE[s][0].b, E,
        tmp16, nullptr, nullptr, nullptr, nullptr, nullptr);
    gemm_wmma_kernel<6, 2><<<grid(E), 256, 0, stream>>>(tmp16, pwPE1[s], pE[s][1].b, E,
        he16, he32, sent, recv, senders, receivers);

    build_nin_kernel<<<N, 3 * LATENT, 0, stream>>>(hn16, sent, recv, in576);
    gemm_wmma_kernel<18, 0><<<grid(N), 256, 0, stream>>>(in576, pwPN0[s], pN[s][0].b, N,
        tmp16, nullptr, nullptr, nullptr, nullptr, nullptr);
    gemm_wmma_kernel<6, 3><<<grid(N), 256, 0, stream>>>(tmp16, pwPN1[s], pN[s][1].b, N,
        hn16, hn32, nullptr, nullptr, nullptr, nullptr);
  }

  // ---- decoder (dec_edge is dead code in the reference; skipped) ----
  gemm_wmma_kernel<6, 0><<<grid(N), 256, 0, stream>>>(hn16, pwD0, dn[0].b, N,
      tmp16, nullptr, nullptr, nullptr, nullptr, nullptr);
  gemm_wmma_kernel<6, 0><<<grid(N), 256, 0, stream>>>(tmp16, pwD1, dn[1].b, N,
      tmp2, nullptr, nullptr, nullptr, nullptr, nullptr);
  dec_final_kernel<<<(N + 7) / 8, 256, 0, stream>>>(tmp2, dn[2].W, dn[2].b, pred, N);

  finalize_kernel<<<(N + 255) / 256, 256, 0, stream>>>(nodes, next_u, pred,
                                                       (float*)d_out, N);
}
